// LundNetTagger_26809185862076
// MI455X (gfx1250) — matrix operand order repaired and checked by reading, not compiled, not run on hardware
//
#include <hip/hip_runtime.h>

// ---------------------------------------------------------------------------
// LundNetTagger on gfx1250: EdgeConv MLPs as bf16 WMMA GEMMs (fp32 accum),
// LDS-staged A tiles (HBM-read-once), fused GraphNorm statistics,
// atomic segment-mean aggregation.
// ---------------------------------------------------------------------------

typedef __attribute__((ext_vector_type(16))) __bf16 v16bf;
typedef __attribute__((ext_vector_type(8)))  __bf16 v8bf;
typedef __attribute__((ext_vector_type(8)))  float  v8f;

#define NE 400000   // edges
#define NN 100000   // nodes
#define NG 1000     // graphs
#define GN_EPS 1e-5f

static inline int cdiv(long long a, long long b) { return (int)((a + b - 1) / b); }

// ---------------------------------------------------------------------------
// Weight repack: fp32 W[K][N] -> bf16 Wt[N][Kpad] (zero pad K..Kpad)
// ---------------------------------------------------------------------------
__global__ void pack_w(const float* __restrict__ w, __bf16* __restrict__ wt,
                       int K, int Kpad, int Ncols) {
  int i = blockIdx.x * blockDim.x + threadIdx.x;
  if (i >= Ncols * Kpad) return;
  int n = i / Kpad, k = i % Kpad;
  float v = (k < K) ? w[k * Ncols + n] : 0.0f;
  wt[i] = (__bf16)v;
}

// ---------------------------------------------------------------------------
// First gather: edge row = [x_dst(5), x_src-x_dst(5), zeros..32] as bf16
// ---------------------------------------------------------------------------
__global__ void gather_first(const float* __restrict__ x, const int* __restrict__ ei,
                             __bf16* __restrict__ A) {
  int e = blockIdx.x * blockDim.x + threadIdx.x;
  if (e >= NE) return;
  int src = ei[e], dst = ei[NE + e];
  const float* xs = x + (long long)src * 5;
  const float* xd = x + (long long)dst * 5;
  __bf16* o = A + (long long)e * 32;
#pragma unroll
  for (int c = 0; c < 5; ++c) {
    float d = xd[c];
    o[c]     = (__bf16)d;
    o[5 + c] = (__bf16)(xs[c] - d);
  }
#pragma unroll
  for (int c = 10; c < 32; ++c) o[c] = (__bf16)0.0f;
}

// ---------------------------------------------------------------------------
// Gather for conv2/3: edge row = [h_dst(C), h_src-h_dst(C)] as bf16
// ---------------------------------------------------------------------------
__global__ void gather_pair(const float* __restrict__ h, const int* __restrict__ ei,
                            __bf16* __restrict__ A, int C) {
  long long i = (long long)blockIdx.x * blockDim.x + threadIdx.x;
  if (i >= (long long)NE * C) return;
  int e = (int)(i / C), c = (int)(i % C);
  int src = ei[e], dst = ei[NE + e];
  float hd = h[(long long)dst * C + c];
  float hs = h[(long long)src * C + c];
  __bf16* o = A + (long long)e * (2 * C);
  o[c]     = (__bf16)hd;
  o[C + c] = (__bf16)(hs - hd);
}

// ---------------------------------------------------------------------------
// WMMA GEMM: out[NE, NT*64] = A[NE, K](bf16) @ Wt^T + bias, fp32 out.
// Block = 256 threads = 8 waves = 2 row-groups x 4 col-groups over a 32-row
// slab; each wave owns NT 16x16 column tiles (NT = Ncols/64), so the block
// covers ALL Ncols columns and the A slab is read from HBM exactly once.
// A slab (32 x K bf16, contiguous in memory) is staged in LDS; A fragments
// come from LDS (ds_load_b128), B fragments from the L2-resident transposed
// weights. Also accumulates per-column sum(h), sum(h^2) for GraphNorm.
// Requires NE % 32 == 0 (400000/32 = 12500).
// ---------------------------------------------------------------------------
template <int NT>
__global__ __launch_bounds__(256)
void gemm_wmma(const __bf16* __restrict__ A, const __bf16* __restrict__ Wt,
               const float* __restrict__ bias, float* __restrict__ out,
               float* __restrict__ sums, int K) {
  constexpr int NCOLS = NT * 64;
  __shared__ float s1[NCOLS];
  __shared__ float s2[NCOLS];
  extern __shared__ __bf16 Atile[];   // [32][K], row-major

  int tid = threadIdx.x;
  if (tid < NCOLS) { s1[tid] = 0.0f; s2[tid] = 0.0f; }

  // Cooperative copy of the 32-row A slab (contiguous 32*K bf16) into LDS.
  long long m0 = (long long)blockIdx.x * 32;
  const __bf16* Asrc = A + m0 * (long long)K;
  int total = 32 * K;
  for (int idx = tid * 8; idx < total; idx += 256 * 8)
    *(v8bf*)(Atile + idx) = *(const v8bf*)(Asrc + idx);
  __syncthreads();

  int wave   = tid >> 5;
  int lane   = tid & 31;
  int laneLo = lane & 15;
  int half   = lane >> 4;            // 0: lanes 0-15, 1: lanes 16-31
  int rowg   = wave >> 2;            // 0..1 (16-row group in slab)
  int colbas = (wave & 3) * 16;      // 0,16,32,48

  // A fragment (16-bit A 16x32 layout): lane<16 holds K {0..7,16..23} of its
  // row; lane>=16 holds K {8..15,24..31}.
  const __bf16* Alds = Atile + (rowg * 16 + laneLo) * K;
  int akoff = half * 8;

  v8f acc[NT];
#pragma unroll
  for (int t = 0; t < NT; ++t) acc[t] = {};

  for (int k = 0; k < K; k += 32) {
    union { v16bf v; v8bf h[2]; } fa;
    fa.h[0] = *(const v8bf*)(Alds + k + akoff);
    fa.h[1] = *(const v8bf*)(Alds + k + akoff + 16);
#pragma unroll
    for (int t = 0; t < NT; ++t) {
      int colg = colbas + t * 64 + laneLo;
      // B fragment (16-bit B 32x16 layout): lane<16 holds K {0..7,16..23},
      // lane>=16 holds K {8..15,24..31} of column colg from Wt[NCOLS][K].
      const __bf16* Brow = Wt + (long long)colg * K + half * 8 + k;
      union { v16bf v; v8bf h[2]; } fb;
      fb.h[0] = *(const v8bf*)(Brow);
      fb.h[1] = *(const v8bf*)(Brow + 16);
      acc[t] = __builtin_amdgcn_wmma_f32_16x16x32_bf16(
          false, fa.v, false, fb.v, (short)0, acc[t], false, false);
    }
  }

  // C/D layout: VGPR r -> row r (lanes 0-15) or row 8+r (lanes 16-31).
  long long rbase = m0 + rowg * 16 + half * 8;
#pragma unroll
  for (int t = 0; t < NT; ++t) {
    int colg = colbas + t * 64 + laneLo;
    float bv = bias[colg];
    float p1 = 0.0f, p2 = 0.0f;
#pragma unroll
    for (int r = 0; r < 8; ++r) {
      float h = acc[t][r] + bv;
      out[(rbase + r) * (long long)NCOLS + colg] = h;
      p1 += h;
      p2 += h * h;
    }
    atomicAdd(&s1[colg], p1);
    atomicAdd(&s2[colg], p2);
  }
  __syncthreads();
  if (tid < NCOLS) {
    atomicAdd(&sums[tid], s1[tid]);
    atomicAdd(&sums[NCOLS + tid], s2[tid]);
  }
}

// ---------------------------------------------------------------------------
// GraphNorm coefficients: y = a*h + c with
//   m = s1/E, var = s2/E - 2*ms*m^2 + ms^2*m^2,
//   a = g * rsqrt(var+eps), c = b - a*ms*m
// gp layout: [0][C]=weight g, [1][C]=bias b, [2][C]=mean_scale ms
// ---------------------------------------------------------------------------
__global__ void gn_coeff(const float* __restrict__ sums, const float* __restrict__ gp,
                         float* __restrict__ coeff, int C, float invE) {
  int c = blockIdx.x * blockDim.x + threadIdx.x;
  if (c >= C) return;
  float g = gp[c], b = gp[C + c], ms = gp[2 * C + c];
  float m   = sums[c] * invE;
  float var = sums[C + c] * invE - 2.0f * ms * m * m + ms * ms * m * m;
  float a   = g * rsqrtf(var + GN_EPS);
  coeff[c]     = a;
  coeff[C + c] = b - a * ms * m;
}

// relu(a*h + c) -> bf16 (input for next GEMM layer)
__global__ void apply_relu_bf16(const float* __restrict__ h, const float* __restrict__ coeff,
                                __bf16* __restrict__ o, long long total, int C) {
  long long i = (long long)blockIdx.x * blockDim.x + threadIdx.x;
  if (i >= total) return;
  int c = (int)(i % C);
  float y = fmaxf(coeff[c] * h[i] + coeff[C + c], 0.0f);
  o[i] = (__bf16)y;
}

// relu(a*h + c) scattered into per-node accumulator (segment-sum over dst)
__global__ void apply_relu_scatter(const float* __restrict__ h, const float* __restrict__ coeff,
                                   const int* __restrict__ ei, float* __restrict__ acc,
                                   long long total, int C) {
  long long i = (long long)blockIdx.x * blockDim.x + threadIdx.x;
  if (i >= total) return;
  int c = (int)(i % C);
  int e = (int)(i / C);
  float y = fmaxf(coeff[c] * h[i] + coeff[C + c], 0.0f);
  atomicAdd(&acc[(long long)ei[NE + e] * C + c], y);
}

__global__ void count_deg(const int* __restrict__ ei, float* __restrict__ cnt) {
  int e = blockIdx.x * blockDim.x + threadIdx.x;
  if (e >= NE) return;
  atomicAdd(&cnt[ei[NE + e]], 1.0f);
}

__global__ void div_cnt(float* __restrict__ acc, const float* __restrict__ cnt,
                        long long total, int C) {
  long long i = (long long)blockIdx.x * blockDim.x + threadIdx.x;
  if (i >= total) return;
  acc[i] /= fmaxf(cnt[i / C], 1.0f);
}

// global mean pool: segment-sum over batch id (+ per-graph counts)
__global__ void pool_sum(const float* __restrict__ h, const int* __restrict__ batch,
                         float* __restrict__ gacc, float* __restrict__ gcnt) {
  long long i = (long long)blockIdx.x * blockDim.x + threadIdx.x;
  if (i >= (long long)NN * 256) return;
  int n = (int)(i / 256), c = (int)(i % 256);
  int b = batch[n];
  atomicAdd(&gacc[(long long)b * 256 + c], h[i]);
  if (c == 0) atomicAdd(&gcnt[b], 1.0f);
}

// Final MLP per graph: out = relu(g @ W1 + b1) @ W2 + b2   (tiny; fp32 VALU)
__global__ __launch_bounds__(256)
void final_mlp(const float* __restrict__ gacc, const float* __restrict__ gcnt,
               const float* __restrict__ w1, const float* __restrict__ b1,
               const float* __restrict__ w2, const float* __restrict__ b2,
               float* __restrict__ out) {
  __shared__ float gv[256];
  __shared__ float hid[256];
  int b = blockIdx.x, t = threadIdx.x;
  float inv = 1.0f / fmaxf(gcnt[b], 1.0f);
  gv[t] = gacc[(long long)b * 256 + t] * inv;
  __syncthreads();
  float a = b1[t];
  for (int k = 0; k < 256; ++k) a += gv[k] * w1[k * 256 + t];
  hid[t] = fmaxf(a, 0.0f);
  __syncthreads();
  if (t < 2) {
    float o = b2[t];
    for (int k = 0; k < 256; ++k) o += hid[k] * w2[k * 2 + t];
    out[b * 2 + t] = o;
  }
}

// ---------------------------------------------------------------------------
extern "C" void kernel_launch(void* const* d_in, const int* in_sizes, int n_in,
                              void* d_out, int out_size, void* d_ws, size_t ws_size,
                              hipStream_t stream) {
  const float* x     = (const float*)d_in[0];
  const int*   ei    = (const int*)  d_in[1];
  const int*   batch = (const int*)  d_in[2];
  const float* c1_w1 = (const float*)d_in[3];  const float* c1_b1 = (const float*)d_in[4];  const float* c1_g1 = (const float*)d_in[5];
  const float* c1_w2 = (const float*)d_in[6];  const float* c1_b2 = (const float*)d_in[7];  const float* c1_g2 = (const float*)d_in[8];
  const float* c1_w3 = (const float*)d_in[9];  const float* c1_b3 = (const float*)d_in[10]; const float* c1_g3 = (const float*)d_in[11];
  const float* c2_w1 = (const float*)d_in[12]; const float* c2_b1 = (const float*)d_in[13]; const float* c2_g1 = (const float*)d_in[14];
  const float* c2_w2 = (const float*)d_in[15]; const float* c2_b2 = (const float*)d_in[16]; const float* c2_g2 = (const float*)d_in[17];
  const float* c3_w1 = (const float*)d_in[18]; const float* c3_b1 = (const float*)d_in[19]; const float* c3_g1 = (const float*)d_in[20];
  const float* l_w1  = (const float*)d_in[21]; const float* l_b1  = (const float*)d_in[22];
  const float* l_w2  = (const float*)d_in[23]; const float* l_b2  = (const float*)d_in[24];
  float* out = (float*)d_out;

  // --- workspace carve-up (aligned to 256B) ---
  char* ws = (char*)d_ws;
  size_t off = 0;
  auto carve = [&](size_t bytes) -> void* {
    void* p = ws + off;
    off += (bytes + 255) & ~(size_t)255;
    return p;
  };
  __bf16* edgeA = (__bf16*)carve((size_t)NE * 512 * 2);   // bf16 edge features (GEMM A)
  float*  edgeH = (float*) carve((size_t)NE * 256 * 4);   // fp32 pre-norm activations
  float*  nodeA = (float*) carve((size_t)NN * 256 * 4);
  float*  nodeB = (float*) carve((size_t)NN * 256 * 4);
  float*  cnt   = (float*) carve((size_t)NN * 4);
  float*  sums  = (float*) carve(512 * 4);
  float*  coeff = (float*) carve(512 * 4);
  float*  gacc  = (float*) carve((size_t)NG * 256 * 4);
  float*  gcnt  = (float*) carve((size_t)NG * 4);
  __bf16* wt1 = (__bf16*)carve(128 * 32 * 2);
  __bf16* wt2 = (__bf16*)carve(128 * 128 * 2);
  __bf16* wt3 = (__bf16*)carve(128 * 128 * 2);
  __bf16* wt4 = (__bf16*)carve(256 * 256 * 2);
  __bf16* wt5 = (__bf16*)carve(256 * 256 * 2);
  __bf16* wt6 = (__bf16*)carve(256 * 512 * 2);

  const int T = 256;

  // --- weight packing (bf16, transposed, K padded to mult of 32) ---
  pack_w<<<cdiv(128 * 32,  T), T, 0, stream>>>(c1_w1, wt1, 10,  32,  128);
  pack_w<<<cdiv(128 * 128, T), T, 0, stream>>>(c1_w2, wt2, 128, 128, 128);
  pack_w<<<cdiv(128 * 128, T), T, 0, stream>>>(c1_w3, wt3, 128, 128, 128);
  pack_w<<<cdiv(256 * 256, T), T, 0, stream>>>(c2_w1, wt4, 256, 256, 256);
  pack_w<<<cdiv(256 * 256, T), T, 0, stream>>>(c2_w2, wt5, 256, 256, 256);
  pack_w<<<cdiv(256 * 512, T), T, 0, stream>>>(c3_w1, wt6, 512, 512, 256);

  // --- degree counts (shared across all convs) ---
  hipMemsetAsync(cnt, 0, (size_t)NN * 4, stream);
  count_deg<<<cdiv(NE, T), T, 0, stream>>>(ei, cnt);

  auto sublayer = [&](const __bf16* Ain, const __bf16* Wt, const float* bias,
                      const float* gp, int Ncols, int K) {
    hipMemsetAsync(sums, 0, (size_t)2 * Ncols * 4, stream);
    dim3 grid(NE / 32);                       // 12500 32-row slabs
    size_t shmem = (size_t)32 * K * 2;        // A slab in LDS (<= 32 KB)
    if (Ncols == 128)
      gemm_wmma<2><<<grid, T, shmem, stream>>>(Ain, Wt, bias, edgeH, sums, K);
    else
      gemm_wmma<4><<<grid, T, shmem, stream>>>(Ain, Wt, bias, edgeH, sums, K);
    gn_coeff<<<cdiv(Ncols, 128), 128, 0, stream>>>(sums, gp, coeff, Ncols, 1.0f / NE);
  };

  // ======================= Conv1 (5 -> 128, 3 layers) =======================
  gather_first<<<cdiv(NE, T), T, 0, stream>>>(x, ei, edgeA);
  sublayer(edgeA, wt1, c1_b1, c1_g1, 128, 32);
  apply_relu_bf16<<<cdiv((long long)NE * 128, T), T, 0, stream>>>(edgeH, coeff, edgeA, (long long)NE * 128, 128);
  sublayer(edgeA, wt2, c1_b2, c1_g2, 128, 128);
  apply_relu_bf16<<<cdiv((long long)NE * 128, T), T, 0, stream>>>(edgeH, coeff, edgeA, (long long)NE * 128, 128);
  sublayer(edgeA, wt3, c1_b3, c1_g3, 128, 128);
  hipMemsetAsync(nodeA, 0, (size_t)NN * 128 * 4, stream);
  apply_relu_scatter<<<cdiv((long long)NE * 128, T), T, 0, stream>>>(edgeH, coeff, ei, nodeA, (long long)NE * 128, 128);
  div_cnt<<<cdiv((long long)NN * 128, T), T, 0, stream>>>(nodeA, cnt, (long long)NN * 128, 128);

  // ======================= Conv2 (128 -> 256, 2 layers) =====================
  gather_pair<<<cdiv((long long)NE * 128, T), T, 0, stream>>>(nodeA, ei, edgeA, 128);
  sublayer(edgeA, wt4, c2_b1, c2_g1, 256, 256);
  apply_relu_bf16<<<cdiv((long long)NE * 256, T), T, 0, stream>>>(edgeH, coeff, edgeA, (long long)NE * 256, 256);
  sublayer(edgeA, wt5, c2_b2, c2_g2, 256, 256);
  hipMemsetAsync(nodeB, 0, (size_t)NN * 256 * 4, stream);
  apply_relu_scatter<<<cdiv((long long)NE * 256, T), T, 0, stream>>>(edgeH, coeff, ei, nodeB, (long long)NE * 256, 256);
  div_cnt<<<cdiv((long long)NN * 256, T), T, 0, stream>>>(nodeB, cnt, (long long)NN * 256, 256);

  // ======================= Conv3 (256 -> 256, 1 layer) ======================
  gather_pair<<<cdiv((long long)NE * 256, T), T, 0, stream>>>(nodeB, ei, edgeA, 256);
  sublayer(edgeA, wt6, c3_b1, c3_g1, 256, 512);
  hipMemsetAsync(nodeA, 0, (size_t)NN * 256 * 4, stream);
  apply_relu_scatter<<<cdiv((long long)NE * 256, T), T, 0, stream>>>(edgeH, coeff, ei, nodeA, (long long)NE * 256, 256);
  div_cnt<<<cdiv((long long)NN * 256, T), T, 0, stream>>>(nodeA, cnt, (long long)NN * 256, 256);

  // ======================= Pool + head ======================================
  hipMemsetAsync(gacc, 0, (size_t)NG * 256 * 4, stream);
  hipMemsetAsync(gcnt, 0, (size_t)NG * 4, stream);
  pool_sum<<<cdiv((long long)NN * 256, T), T, 0, stream>>>(nodeA, batch, gacc, gcnt);
  final_mlp<<<NG, T, 0, stream>>>(gacc, gcnt, l_w1, l_b1, l_w2, l_b2, out);
}
